// MultiLevelFeatureSampler_52106543235142
// MI455X (gfx1250) — matrix operand order repaired and checked by reading, not compiled, not run on hardware
//
#include <hip/hip_runtime.h>
#include <hip/hip_bf16.h>

typedef __attribute__((ext_vector_type(16))) __bf16 v16bf;
typedef __attribute__((ext_vector_type(8)))  __bf16 v8bf;
typedef __attribute__((ext_vector_type(8)))  float  v8f;

union V16 { v16bf v; v8bf h[2]; };

#define LDKA 104          // A row stride (bf16): 208B/row, 16B aligned, conflict-free
#define LDKW 104          // W row stride (bf16): 208B/row, 16B aligned, conflict-free
#define NPTS 512
#define CCH  128
#define OUTC 256
#define DTOT 83

// LDS layout (bytes) — hot W regions first so all B-fragment ds_loads use
// 16-bit immediate offsets from one per-tile base VGPR (Wlo delta = 53248).
#define OFF_WHI 0                        // 256*104*2 = 53248
#define OFF_WLO 53248                    // 53248
#define OFF_AHI 106496                   // 128*104*2 = 26624
#define OFF_ALO 133120                   // 26624
#define OFF_IDX 159744                   // 83*4 -> pad 512
#define SMEM_BYTES 160256                // 2 workgroups fit in a 320KB WGP

__global__ __launch_bounds__(256) void mlfs_wmma_kernel(
    const float* __restrict__ points,
    const float* __restrict__ feat0,
    const float* __restrict__ feat1,
    const float* __restrict__ feat2,
    const float* __restrict__ Wm,
    const float* __restrict__ bias,
    float* __restrict__ out)
{
  extern __shared__ char smem[];
  __bf16* Whi = (__bf16*)(smem + OFF_WHI);
  __bf16* Wlo = (__bf16*)(smem + OFF_WLO);
  __bf16* Ahi = (__bf16*)(smem + OFF_AHI);
  __bf16* Alo = (__bf16*)(smem + OFF_ALO);
  int*    idx = (int*)   (smem + OFF_IDX);

  const int tid = threadIdx.x;
  const int blk = blockIdx.x;
  const int b_  = blk >> 9;          // / 512
  const int p   = blk & 511;

  // ---- per-point sample index table (83 entries) --------------------
  if (tid < DTOT) {
    const float px = points[(b_ * NPTS + p) * 2 + 0];
    const float py = points[(b_ * NPTS + p) * 2 + 1];
    int d = tid, ks, Wd, H, dl;
    if (d < 49)      { ks = 7; Wd = 305; H = 93; dl = d; }
    else if (d < 74) { ks = 5; Wd = 153; H = 47; dl = d - 49; }
    else             { ks = 3; Wd = 77;  H = 24; dl = d - 74; }
    const int half = ks >> 1;
    const int jj = dl / ks, kk = dl % ks;
    const float jo = (float)(jj - half), ko = (float)(kk - half);
    const float x = fminf(fmaxf(px * (float)(Wd - 1), 0.f), (float)(Wd - 1));
    const float y = fminf(fmaxf(py * (float)(H  - 1), 0.f), (float)(H  - 1));
    const int ox = (int)floorf(fminf(fmaxf(x + ko, 0.f), (float)(Wd - 1)));
    const int oy = (int)floorf(fminf(fmaxf(y + jo, 0.f), (float)(H  - 1)));
    idx[d] = oy * Wd + ox;
  }
  __syncthreads();

  // ---- gather patch A(128x83), split fp32 -> bf16 hi/lo -------------
  for (int i = tid; i < CCH * DTOT; i += 256) {
    const int c = i / DTOT;
    const int d = i - c * DTOT;
    const float* f; int hw;
    if (d < 49)      { f = feat0; hw = 93 * 305; }
    else if (d < 74) { f = feat1; hw = 47 * 153; }
    else             { f = feat2; hw = 24 * 77;  }
    const float v = f[(size_t)(b_ * CCH + c) * hw + idx[d]];
    const __bf16 vh = (__bf16)v;
    const __bf16 vl = (__bf16)(v - (float)vh);
    Ahi[c * LDKA + d] = vh;
    Alo[c * LDKA + d] = vl;
  }
  for (int i = tid; i < CCH * (LDKA - DTOT); i += 256) {   // zero K-pad 83..103
    const int c = i / (LDKA - DTOT);
    const int d = DTOT + (i - c * (LDKA - DTOT));
    Ahi[c * LDKA + d] = (__bf16)0.f;
    Alo[c * LDKA + d] = (__bf16)0.f;
  }

  // ---- split W(256x83) -> bf16 hi/lo, zero K-pad to 104 -------------
  for (int i = tid; i < OUTC * DTOT; i += 256) {
    const int r = i / DTOT;
    const int d = i - r * DTOT;
    const float v = Wm[r * DTOT + d];
    const __bf16 vh = (__bf16)v;
    const __bf16 vl = (__bf16)(v - (float)vh);
    Whi[r * LDKW + d] = vh;
    Wlo[r * LDKW + d] = vl;
  }
  for (int i = tid; i < OUTC * (LDKW - DTOT); i += 256) {
    const int r = i / (LDKW - DTOT);
    const int d = DTOT + (i - r * (LDKW - DTOT));
    Whi[r * LDKW + d] = (__bf16)0.f;
    Wlo[r * LDKW + d] = (__bf16)0.f;
  }
  __syncthreads();

  // ---- WMMA compute: wave w owns M-tile w (channels w*16..w*16+15) --
  const int wave = tid >> 5;
  const int lane = tid & 31;
  const int sub  = lane & 15;      // M (A), N (B/D) within tile
  const int hi16 = lane >> 4;      // K half-group selector

  // Preload this wave's A fragments: 3 K-blocks x (hi,lo).
  // K order used here (contiguous, same as B): lanes 0-15 K = kb*32+0..15 ; lanes 16-31 K = kb*32+16..31
  V16 a_hi[3], a_lo[3];
  {
    const __bf16* rH = Ahi + (wave * 16 + sub) * LDKA + hi16 * 16;
    const __bf16* rL = Alo + (wave * 16 + sub) * LDKA + hi16 * 16;
#pragma unroll
    for (int kb = 0; kb < 3; ++kb) {
      a_hi[kb].h[0] = *(const v8bf*)(rH + kb * 32);
      a_hi[kb].h[1] = *(const v8bf*)(rH + kb * 32 + 8);
      a_lo[kb].h[0] = *(const v8bf*)(rL + kb * 32);
      a_lo[kb].h[1] = *(const v8bf*)(rL + kb * 32 + 8);
    }
  }

  // Preload per-lane bias values for all 16 N-tiles.
  float bias_r[16];
#pragma unroll
  for (int nt = 0; nt < 16; ++nt) bias_r[nt] = bias[nt * 16 + sub];

  // Double-buffered B fragments (hi/lo per K-block).
  // K order used here (contiguous, same as A): lanes 0-15 K = kb*32+0..15 ; lanes 16-31 K = kb*32+16..31
  V16 bH[2][3], bL[2][3];
  {
    const __bf16* pH = Whi + sub * LDKW + hi16 * 16;   // nt = 0
    const __bf16* pL = Wlo + sub * LDKW + hi16 * 16;
#pragma unroll
    for (int kb = 0; kb < 3; ++kb) {
      bH[0][kb].h[0] = *(const v8bf*)(pH + kb * 32);
      bH[0][kb].h[1] = *(const v8bf*)(pH + kb * 32 + 8);
      bL[0][kb].h[0] = *(const v8bf*)(pL + kb * 32);
      bL[0][kb].h[1] = *(const v8bf*)(pL + kb * 32 + 8);
    }
  }

  const size_t obase = (size_t)(b_ * NPTS + p) * CCH * OUTC;
  float* orow = out + obase + (size_t)(wave * 16 + hi16 * 8) * OUTC + sub;

#pragma unroll
  for (int nt = 0; nt < 16; ++nt) {
    const int cur = nt & 1;
    // Prefetch next N-tile's 6 B fragments into the other buffer.
    if (nt < 15) {
      const int nxt = cur ^ 1;
      const __bf16* pH = Whi + ((nt + 1) * 16 + sub) * LDKW + hi16 * 16;
      const __bf16* pL = Wlo + ((nt + 1) * 16 + sub) * LDKW + hi16 * 16;
#pragma unroll
      for (int kb = 0; kb < 3; ++kb) {
        bH[nxt][kb].h[0] = *(const v8bf*)(pH + kb * 32);
        bH[nxt][kb].h[1] = *(const v8bf*)(pH + kb * 32 + 8);
        bL[nxt][kb].h[0] = *(const v8bf*)(pL + kb * 32);
        bL[nxt][kb].h[1] = *(const v8bf*)(pL + kb * 32 + 8);
      }
    }
    v8f acc = {};
#pragma unroll
    for (int kb = 0; kb < 3; ++kb) {
      // split-bf16 fp32 emulation: Ah*Bh + Al*Bh + Ah*Bl, fp32 accumulate
      acc = __builtin_amdgcn_wmma_f32_16x16x32_bf16(false, a_hi[kb].v, false, bH[cur][kb].v,
                                                    (short)0, acc, false, false);
      acc = __builtin_amdgcn_wmma_f32_16x16x32_bf16(false, a_lo[kb].v, false, bH[cur][kb].v,
                                                    (short)0, acc, false, false);
      acc = __builtin_amdgcn_wmma_f32_16x16x32_bf16(false, a_hi[kb].v, false, bL[cur][kb].v,
                                                    (short)0, acc, false, false);
    }
    // D layout: elem r -> M = r + hi16*8, N = sub.
    // Output is streamed 256MB, never re-read: store non-temporal so the
    // feature pyramids (~76MB) stay resident in the 192MB L2 for gathers.
    float* op = orow + nt * 16;
#pragma unroll
    for (int r = 0; r < 8; ++r)
      __builtin_nontemporal_store(acc[r] + bias_r[nt], op + (size_t)r * OUTC);
  }
}

extern "C" void kernel_launch(void* const* d_in, const int* in_sizes, int n_in,
                              void* d_out, int out_size, void* d_ws, size_t ws_size,
                              hipStream_t stream) {
  (void)in_sizes; (void)n_in; (void)out_size; (void)d_ws; (void)ws_size;
  const float* points = (const float*)d_in[0];
  const float* feat0  = (const float*)d_in[1];
  const float* feat1  = (const float*)d_in[2];
  const float* feat2  = (const float*)d_in[3];
  const float* Wm     = (const float*)d_in[4];
  const float* bias   = (const float*)d_in[5];
  float* out = (float*)d_out;

  dim3 grid(4 * NPTS);   // one workgroup per (batch, point)
  dim3 block(256);       // 8 wave32s
  mlfs_wmma_kernel<<<grid, block, SMEM_BYTES, stream>>>(
      points, feat0, feat1, feat2, Wm, bias, out);
}
